// PosteriorDecoder_Qz_25151328485519
// MI455X (gfx1250) — matrix operand order repaired and checked by reading, not compiled, not run on hardware
//
#include <hip/hip_runtime.h>
#include <hip/hip_bf16.h>
#include <math.h>

// ---------------- problem constants ----------------
#define Bv   16
#define TUv  128
#define TMv  128
#define TZv  64
#define Tv   320            // TU+TM+TZ
#define Vv   32000
#define Ev   512
#define Hv   512
#define Sv   1024           // E+H
#define PW   (Vv + Tv)      // 32320 softmax width
#define NCH_ATT 4           // N-chunks for attention GEMM (32 n-tiles / 8)
#define NCH_CP  8           // N-chunks for copy GEMM      (64 n-tiles / 8)

typedef __attribute__((ext_vector_type(16))) __bf16 v16bf;
typedef __attribute__((ext_vector_type(8)))  float  v8f;

// ---------------- WMMA tile loaders (contiguous k = 16*(lane/16)+i for A and B, wave32) ----------
__device__ inline v16bf load_a16x32(const __bf16* __restrict__ A, int lda, int lane) {
  int row = lane & 15;
  int kb  = (lane >> 4) * 16;
  const __bf16* p = A + (size_t)row * lda + kb;
  v16bf a;
#pragma unroll
  for (int i = 0; i < 16; ++i) a[i] = p[i];
  return a;
}
__device__ inline v16bf load_bT32x16(const __bf16* __restrict__ Bt, int ldb, int lane) {
  int n  = lane & 15;
  int kb = (lane >> 4) * 16;
  const __bf16* p = Bt + (size_t)n * ldb + kb;
  v16bf b;
#pragma unroll
  for (int i = 0; i < 16; ++i) b[i] = p[i];
  return b;
}

__device__ inline float sigmoidf_(float x) { return 1.f / (1.f + expf(-x)); }

// ---------------- conversions -------------------------------------------
__global__ void k_conv_enc(const float* __restrict__ uh, const float* __restrict__ mh,
                           const float* __restrict__ pzh, __bf16* __restrict__ encb) {
  int idx = blockIdx.x * blockDim.x + threadIdx.x;            // < 5120*512
  int row = idx >> 9, k = idx & 511;
  int b = row / Tv, t = row % Tv;
  float v;
  if (t < TUv)        v = uh[((size_t)b * TUv + t) * Hv + k];
  else if (t < 2*TUv) v = mh[((size_t)b * TMv + (t - TUv)) * Hv + k];
  else                v = pzh[((size_t)b * TZv + (t - 2*TUv)) * Hv + k];
  encb[idx] = (__bf16)v;
}
__global__ void k_conv_webt(const float* __restrict__ Wattn, __bf16* __restrict__ webt) {
  int idx = blockIdx.x * blockDim.x + threadIdx.x;            // < 512*512
  int n = idx >> 9, k = idx & 511;
  webt[idx] = (__bf16)Wattn[(size_t)n * (2 * Hv) + Hv + k];
}
__global__ void k_conv_f32_bf16(const float* __restrict__ src, __bf16* __restrict__ dst, int n) {
  int idx = blockIdx.x * blockDim.x + threadIdx.x;
  if (idx < n) dst[idx] = (__bf16)src[idx];
}

// hq_proj: wave-per-j GEMV, coalesced (uses W_attn[:, 0:512], row stride 2H)
__global__ void k_hqproj(const float* __restrict__ Wattn, const float* __restrict__ battn,
                         const float* __restrict__ h0, float* __restrict__ hq) {
  __shared__ float sh[Bv * Hv];                 // 32 KB
  for (int i = threadIdx.x; i < Bv * Hv; i += blockDim.x) sh[i] = h0[i];
  __syncthreads();
  int lane = threadIdx.x & 31;
  int j = (blockIdx.x * blockDim.x + threadIdx.x) >> 5;       // 0..511
  const float* wr = Wattn + (size_t)j * (2 * Hv);
  float acc[Bv];
#pragma unroll
  for (int b = 0; b < Bv; ++b) acc[b] = 0.f;
  for (int k0 = 0; k0 < Hv; k0 += 32) {
    float wv = wr[k0 + lane];
#pragma unroll
    for (int b = 0; b < Bv; ++b) acc[b] += wv * sh[b * Hv + k0 + lane];
  }
#pragma unroll
  for (int b = 0; b < Bv; ++b)
    for (int off = 16; off > 0; off >>= 1) acc[b] += __shfl_xor(acc[b], off, 32);
  if (lane == 0) {
    float bb = battn[j];
#pragma unroll
    for (int b = 0; b < Bv; ++b) hq[b * Hv + j] = acc[b] + bb;
  }
}

// ---------------- fused attention GEMM (WMMA bf16, N split in 4) ----------
// part[c*5120 + m] = sum_{n in chunk c} tanh(hq[b,n] + enc[m] @ We^T[:,n]) * v_attn[n]
__global__ void k_attn_gemm(const __bf16* __restrict__ encb, const __bf16* __restrict__ webt,
                            const float* __restrict__ hq, const float* __restrict__ vattn,
                            float* __restrict__ part) {
  int lane = threadIdx.x & 31;
  int w = (blockIdx.x * blockDim.x + threadIdx.x) >> 5;       // 0..1279
  int chunk = w / (Bv * Tv / 16);                             // 0..3
  int tile  = w % (Bv * Tv / 16);                             // 0..319
  int m0 = tile * 16;
  int b  = m0 / Tv;
  int nl = lane & 15;
  float acc[8];
#pragma unroll
  for (int r = 0; r < 8; ++r) acc[r] = 0.f;
  for (int ntl = 0; ntl < 8; ++ntl) {
    int nt = chunk * 8 + ntl;
    v8f c = {};
    for (int kt = 0; kt < Hv / 32; ++kt) {
      v16bf a  = load_a16x32(encb + (size_t)m0 * Hv + kt * 32, Hv, lane);
      v16bf bm = load_bT32x16(webt + (size_t)(nt * 16) * Hv + kt * 32, Hv, lane);
      c = __builtin_amdgcn_wmma_f32_16x16x32_bf16(false, a, false, bm, (short)0, c, false, false);
    }
    int ng = nt * 16 + nl;
    float h = hq[b * Hv + ng], va = vattn[ng];
#pragma unroll
    for (int r = 0; r < 8; ++r) acc[r] += tanhf(c[r] + h) * va;
  }
#pragma unroll
  for (int r = 0; r < 8; ++r) {
    acc[r] += __shfl_xor(acc[r], 1, 32);
    acc[r] += __shfl_xor(acc[r], 2, 32);
    acc[r] += __shfl_xor(acc[r], 4, 32);
    acc[r] += __shfl_xor(acc[r], 8, 32);
  }
  float* dst = part + (size_t)chunk * (Bv * Tv) + m0;
  if (lane == 0) {
#pragma unroll
    for (int r = 0; r < 8; ++r) dst[r] = acc[r];
  } else if (lane == 16) {
#pragma unroll
    for (int r = 0; r < 8; ++r) dst[8 + r] = acc[r];
  }
}
__global__ void k_att_reduce(const float* __restrict__ part, float* __restrict__ attsc) {
  int m = blockIdx.x * blockDim.x + threadIdx.x;              // < 5120
  float s = 0.f;
#pragma unroll
  for (int c = 0; c < NCH_ATT; ++c) s += part[(size_t)c * (Bv * Tv) + m];
  attsc[m] = s;
}

// ---------------- attention softmax + context -----------------------------
__global__ void k_attn_ctx(const float* __restrict__ attsc, const float* __restrict__ uh,
                           const float* __restrict__ mh, const float* __restrict__ pzh,
                           float* __restrict__ ctx) {
  __shared__ float sw[Tv];
  __shared__ float red[256];
  int b = blockIdx.x, tid = threadIdx.x;
  float lm = -3.4e38f;
  for (int i = tid; i < Tv; i += 256) { sw[i] = attsc[b * Tv + i]; lm = fmaxf(lm, sw[i]); }
  red[tid] = lm; __syncthreads();
  for (int s = 128; s > 0; s >>= 1) { if (tid < s) red[tid] = fmaxf(red[tid], red[tid + s]); __syncthreads(); }
  float mx = red[0]; __syncthreads();
  float ls = 0.f;
  for (int i = tid; i < Tv; i += 256) { float e = expf(sw[i] - mx); sw[i] = e; ls += e; }
  red[tid] = ls; __syncthreads();
  for (int s = 128; s > 0; s >>= 1) { if (tid < s) red[tid] += red[tid + s]; __syncthreads(); }
  float inv = 1.f / red[0]; __syncthreads();
  for (int i = tid; i < Tv; i += 256) sw[i] *= inv;
  __syncthreads();
  for (int h = tid; h < Hv; h += 256) {
    float acc = 0.f;
    for (int t = 0; t < TUv; ++t) acc += sw[t]         * uh[((size_t)b * TUv + t) * Hv + h];
    for (int t = 0; t < TMv; ++t) acc += sw[TUv + t]   * mh[((size_t)b * TMv + t) * Hv + h];
    for (int t = 0; t < TZv; ++t) acc += sw[2*TUv + t] * pzh[((size_t)b * TZv + t) * Hv + h];
    ctx[b * Hv + h] = acc;
  }
}

// ---------------- GRU matvecs ---------------------------------------------
__global__ void k_gemv_gx(const float* __restrict__ Wih, const float* __restrict__ bih,
                          const float* __restrict__ emb, const float* __restrict__ ctx,
                          float* __restrict__ gx) {
  __shared__ float sx[Bv * Sv];                    // 64 KB
  for (int i = threadIdx.x; i < Bv * Sv; i += blockDim.x) {
    int b = i >> 10, k = i & 1023;
    sx[i] = (k < Ev) ? emb[b * Ev + k] : ctx[b * Hv + (k - Ev)];
  }
  __syncthreads();
  int lane = threadIdx.x & 31;
  int j = (blockIdx.x * blockDim.x + threadIdx.x) >> 5;       // 0..1535
  const float* wr = Wih + (size_t)j * Sv;
  float acc[Bv];
#pragma unroll
  for (int b = 0; b < Bv; ++b) acc[b] = 0.f;
  for (int k0 = 0; k0 < Sv; k0 += 32) {
    float wv = wr[k0 + lane];
#pragma unroll
    for (int b = 0; b < Bv; ++b) acc[b] += wv * sx[b * Sv + k0 + lane];
  }
#pragma unroll
  for (int b = 0; b < Bv; ++b)
    for (int off = 16; off > 0; off >>= 1) acc[b] += __shfl_xor(acc[b], off, 32);
  if (lane == 0) {
    float bb = bih[j];
#pragma unroll
    for (int b = 0; b < Bv; ++b) gx[b * 1536 + j] = acc[b] + bb;
  }
}
__global__ void k_gemv_gh(const float* __restrict__ Whh, const float* __restrict__ bhh,
                          const float* __restrict__ h0, float* __restrict__ gh) {
  __shared__ float sh[Bv * Hv];                    // 32 KB
  for (int i = threadIdx.x; i < Bv * Hv; i += blockDim.x) sh[i] = h0[i];
  __syncthreads();
  int lane = threadIdx.x & 31;
  int j = (blockIdx.x * blockDim.x + threadIdx.x) >> 5;
  const float* wr = Whh + (size_t)j * Hv;
  float acc[Bv];
#pragma unroll
  for (int b = 0; b < Bv; ++b) acc[b] = 0.f;
  for (int k0 = 0; k0 < Hv; k0 += 32) {
    float wv = wr[k0 + lane];
#pragma unroll
    for (int b = 0; b < Bv; ++b) acc[b] += wv * sh[b * Hv + k0 + lane];
  }
#pragma unroll
  for (int b = 0; b < Bv; ++b)
    for (int off = 16; off > 0; off >>= 1) acc[b] += __shfl_xor(acc[b], off, 32);
  if (lane == 0) {
    float bb = bhh[j];
#pragma unroll
    for (int b = 0; b < Bv; ++b) gh[b * 1536 + j] = acc[b] + bb;
  }
}

// ---------------- GRU cell -------------------------------------------------
__global__ void k_gru_cell(const float* __restrict__ gx, const float* __restrict__ gh,
                           const float* __restrict__ h0, const float* __restrict__ emb,
                           float* __restrict__ out, float* __restrict__ st) {
  int idx = blockIdx.x * blockDim.x + threadIdx.x;            // < 16*512
  int b = idx >> 9, j = idx & 511;
  float xr = gx[b * 1536 + j], xz = gx[b * 1536 + 512 + j], xn = gx[b * 1536 + 1024 + j];
  float hr = gh[b * 1536 + j], hz = gh[b * 1536 + 512 + j], hn = gh[b * 1536 + 1024 + j];
  float r = sigmoidf_(xr + hr);
  float z = sigmoidf_(xz + hz);
  float n = tanhf(xn + r * hn);
  float hv = h0[b * Hv + j];
  float hnew = (1.f - z) * n + z * hv;
  out[(size_t)Bv * Vv + b * Hv + j]           = hnew;   // hnew (1,B,H)
  out[(size_t)Bv * Vv + Bv * Hv + b * Hv + j] = hnew;   // gru_out (B,1,H)
  st[b * Sv + j]      = hnew;
  st[b * Sv + Hv + j] = emb[b * Ev + j];
}

// ---------------- score_g: fp32 GEMV over Wgen (131 MB, read once) --------
// st staged into LDS via CDNA5 async global->LDS loads (ASYNCcnt tracked).
__global__ void k_score_gen(const float* __restrict__ Wgen, const float* __restrict__ bgen,
                            const float* __restrict__ st, float* __restrict__ logits) {
  __shared__ float sst[Bv * Sv];                  // 64 KB
  for (int i = threadIdx.x * 4; i < Bv * Sv; i += blockDim.x * 4) {
    unsigned lds_a = (unsigned)(size_t)&sst[i];
    const float* gp = st + i;
    asm volatile("global_load_async_to_lds_b128 %0, %1, off"
                 :: "v"(lds_a), "v"(gp) : "memory");
  }
  asm volatile("s_wait_asynccnt 0" ::: "memory");
  __syncthreads();
  int lane = threadIdx.x & 31;
  int n = (blockIdx.x * blockDim.x + threadIdx.x) >> 5;       // 0..31999
  const float* wr = Wgen + (size_t)n * Sv;
  float acc[Bv];
#pragma unroll
  for (int b = 0; b < Bv; ++b) acc[b] = 0.f;
  for (int k0 = 0; k0 < Sv; k0 += 32) {
    float wv = wr[k0 + lane];
#pragma unroll
    for (int b = 0; b < Bv; ++b) acc[b] += wv * sst[b * Sv + k0 + lane];
  }
#pragma unroll
  for (int b = 0; b < Bv; ++b)
    for (int off = 16; off > 0; off >>= 1) acc[b] += __shfl_xor(acc[b], off, 32);
  if (lane == 0) {
    float bb = bgen[n];
#pragma unroll
    for (int b = 0; b < Bv; ++b) logits[(size_t)b * PW + n] = acc[b] + bb;
  }
}

// ---------------- fused copy-score GEMMs (WMMA bf16, N split in 8) --------
__global__ void k_copy_gemm(const __bf16* __restrict__ encb,
                            const __bf16* __restrict__ wu, const __bf16* __restrict__ wm,
                            const __bf16* __restrict__ wz,
                            const float* __restrict__ bu, const float* __restrict__ bm_,
                            const float* __restrict__ bz,
                            const float* __restrict__ st, float* __restrict__ part) {
  int lane = threadIdx.x & 31;
  int w = (blockIdx.x * blockDim.x + threadIdx.x) >> 5;       // 0..2559
  int chunk = w / (Bv * Tv / 16);                             // 0..7
  int tile  = w % (Bv * Tv / 16);                             // 0..319
  int m0 = tile * 16;
  int b  = m0 / Tv;
  int t0 = m0 % Tv;
  const __bf16* Wt; const float* bias;
  if (t0 < TUv)        { Wt = wu; bias = bu;  }
  else if (t0 < 2*TUv) { Wt = wm; bias = bm_; }
  else                 { Wt = wz; bias = bz;  }
  int nl = lane & 15;
  float acc[8];
#pragma unroll
  for (int r = 0; r < 8; ++r) acc[r] = 0.f;
  for (int ntl = 0; ntl < 8; ++ntl) {                         // 8 of 64 N-tiles
    int nt = chunk * 8 + ntl;
    v8f c = {};
    for (int kt = 0; kt < Hv / 32; ++kt) {                    // 16 K-steps
      v16bf a  = load_a16x32(encb + (size_t)m0 * Hv + kt * 32, Hv, lane);
      v16bf bt = load_bT32x16(Wt + (size_t)(nt * 16) * Hv + kt * 32, Hv, lane);
      c = __builtin_amdgcn_wmma_f32_16x16x32_bf16(false, a, false, bt, (short)0, c, false, false);
    }
    int ng = nt * 16 + nl;
    float bb = bias[ng], sv = st[b * Sv + ng];
#pragma unroll
    for (int r = 0; r < 8; ++r) acc[r] += tanhf(c[r] + bb) * sv;
  }
#pragma unroll
  for (int r = 0; r < 8; ++r) {
    acc[r] += __shfl_xor(acc[r], 1, 32);
    acc[r] += __shfl_xor(acc[r], 2, 32);
    acc[r] += __shfl_xor(acc[r], 4, 32);
    acc[r] += __shfl_xor(acc[r], 8, 32);
  }
  float* dst = part + (size_t)chunk * (Bv * Tv) + m0;
  if (lane == 0) {
#pragma unroll
    for (int r = 0; r < 8; ++r) dst[r] = acc[r];
  } else if (lane == 16) {
#pragma unroll
    for (int r = 0; r < 8; ++r) dst[8 + r] = acc[r];
  }
}
__global__ void k_cp_reduce(const float* __restrict__ part, float* __restrict__ logits) {
  int m = blockIdx.x * blockDim.x + threadIdx.x;              // < 5120
  float s = 0.f;
#pragma unroll
  for (int c = 0; c < NCH_CP; ++c) s += part[(size_t)c * (Bv * Tv) + m];
  int b = m / Tv, t = m % Tv;
  logits[(size_t)b * PW + Vv + t] = s;
}

// ---------------- big softmax over 32320 per batch row --------------------
__global__ void k_softmax_big(float* __restrict__ logits) {
  __shared__ float red[256];
  int b = blockIdx.x, tid = threadIdx.x;
  float* base = logits + (size_t)b * PW;
  float lm = -3.4e38f;
  for (int i = tid; i < PW; i += 256) lm = fmaxf(lm, base[i]);
  red[tid] = lm; __syncthreads();
  for (int s = 128; s > 0; s >>= 1) { if (tid < s) red[tid] = fmaxf(red[tid], red[tid + s]); __syncthreads(); }
  float mx = red[0]; __syncthreads();
  float ls = 0.f;
  for (int i = tid; i < PW; i += 256) ls += expf(base[i] - mx);
  red[tid] = ls; __syncthreads();
  for (int s = 128; s > 0; s >>= 1) { if (tid < s) red[tid] += red[tid + s]; __syncthreads(); }
  float inv = 1.f / red[0];
  __syncthreads();
  for (int i = tid; i < PW; i += 256) base[i] = expf(base[i] - mx) * inv;
}

// ---------------- prob_out = prob_g + dense pz mix (float4) ---------------
__global__ void k_prob_out(const float* __restrict__ probs, const float* __restrict__ pvz,
                           float* __restrict__ out) {
  __shared__ float spz[TZv];
  int b = blockIdx.y, tid = threadIdx.x;
  if (tid < TZv) spz[tid] = probs[(size_t)b * PW + Vv + 2 * TUv + tid];
  __syncthreads();
  int v4 = blockIdx.x * 256 + tid;                            // 0..7999
  if (v4 >= Vv / 4) return;
  int v = v4 * 4;
  float4 acc = *(const float4*)(probs + (size_t)b * PW + v);
  const float* pzb = pvz + (size_t)b * TZv * Vv + v;
#pragma unroll 4
  for (int t = 0; t < TZv; ++t) {
    float4 p = *(const float4*)(pzb + (size_t)t * Vv);
    float s = spz[t];
    acc.x += s * p.x; acc.y += s * p.y; acc.z += s * p.z; acc.w += s * p.w;
  }
  *(float4*)(out + (size_t)b * Vv + v) = acc;
}

// ---------------- deterministic scatter of u/m copy probs -----------------
__global__ void k_scatter(const float* __restrict__ probs, const int* __restrict__ ui,
                          const int* __restrict__ mi, float* __restrict__ out) {
  int b = blockIdx.x * blockDim.x + threadIdx.x;
  if (b >= Bv) return;
  float* ob = out + (size_t)b * Vv;
  const float* pb = probs + (size_t)b * PW + Vv;
  for (int t = 0; t < TUv; ++t) ob[ui[b * TUv + t]] += pb[t];
  for (int t = 0; t < TMv; ++t) ob[mi[b * TMv + t]] += pb[TUv + t];
}

// ======================= host launch ======================================
extern "C" void kernel_launch(void* const* d_in, const int* in_sizes, int n_in,
                              void* d_out, int out_size, void* d_ws, size_t ws_size,
                              hipStream_t stream) {
  const int*   u_input  = (const int*)  d_in[0];
  const float* u_hid    = (const float*)d_in[2];
  const int*   m_input  = (const int*)  d_in[3];
  const float* m_hid    = (const float*)d_in[5];
  const float* pvz_prob = (const float*)d_in[6];
  const float* pvz_hid  = (const float*)d_in[7];
  const float* emb_zt   = (const float*)d_in[9];
  const float* last_h   = (const float*)d_in[10];
  const float* W_attn   = (const float*)d_in[11];
  const float* b_attn   = (const float*)d_in[12];
  const float* v_attn   = (const float*)d_in[13];
  const float* W_ih     = (const float*)d_in[14];
  const float* W_hh     = (const float*)d_in[15];
  const float* b_ih     = (const float*)d_in[16];
  const float* b_hh     = (const float*)d_in[17];
  const float* Wgen_w   = (const float*)d_in[18];
  const float* Wgen_b   = (const float*)d_in[19];
  const float* Wcpu_w   = (const float*)d_in[20];
  const float* Wcpu_b   = (const float*)d_in[21];
  const float* Wcpm_w   = (const float*)d_in[22];
  const float* Wcpm_b   = (const float*)d_in[23];
  const float* Wcppz_w  = (const float*)d_in[24];
  const float* Wcppz_b  = (const float*)d_in[25];
  float* out = (float*)d_out;

  char* wp = (char*)d_ws;
  auto carve = [&](size_t bytes) { void* p = (void*)wp; wp += (bytes + 255) & ~(size_t)255; return p; };
  __bf16* encb    = (__bf16*)carve((size_t)Bv * Tv * Hv * 2);
  __bf16* webt    = (__bf16*)carve((size_t)Hv * Hv * 2);
  __bf16* wcpub   = (__bf16*)carve((size_t)Sv * Hv * 2);
  __bf16* wcpmb   = (__bf16*)carve((size_t)Sv * Hv * 2);
  __bf16* wcpzb   = (__bf16*)carve((size_t)Sv * Hv * 2);
  float*  hqproj  = (float*)carve((size_t)Bv * Hv * 4);
  float*  attsc   = (float*)carve((size_t)Bv * Tv * 4);
  float*  ctx     = (float*)carve((size_t)Bv * Hv * 4);
  float*  gx      = (float*)carve((size_t)Bv * 1536 * 4);
  float*  gh      = (float*)carve((size_t)Bv * 1536 * 4);
  float*  st      = (float*)carve((size_t)Bv * Sv * 4);
  float*  logits  = (float*)carve((size_t)Bv * PW * 4);
  float*  attpart = (float*)carve((size_t)NCH_ATT * Bv * Tv * 4);
  float*  cppart  = (float*)carve((size_t)NCH_CP  * Bv * Tv * 4);

  // 1) conversions + hq projection
  k_conv_enc<<<(Bv * Tv * Hv) / 256, 256, 0, stream>>>(u_hid, m_hid, pvz_hid, encb);
  k_conv_webt<<<(Hv * Hv) / 256, 256, 0, stream>>>(W_attn, webt);
  k_conv_f32_bf16<<<(Sv * Hv) / 256, 256, 0, stream>>>(Wcpu_w, wcpub, Sv * Hv);
  k_conv_f32_bf16<<<(Sv * Hv) / 256, 256, 0, stream>>>(Wcpm_w, wcpmb, Sv * Hv);
  k_conv_f32_bf16<<<(Sv * Hv) / 256, 256, 0, stream>>>(Wcppz_w, wcpzb, Sv * Hv);
  k_hqproj<<<Hv / 8, 256, 0, stream>>>(W_attn, b_attn, last_h, hqproj);

  // 2) attention (WMMA, 1280 waves) -> reduce -> softmax+context
  k_attn_gemm<<<(NCH_ATT * Bv * Tv / 16) / 4, 128, 0, stream>>>(encb, webt, hqproj, v_attn, attpart);
  k_att_reduce<<<(Bv * Tv) / 256, 256, 0, stream>>>(attpart, attsc);
  k_attn_ctx<<<Bv, 256, 0, stream>>>(attsc, u_hid, m_hid, pvz_hid, ctx);

  // 3) GRU
  k_gemv_gx<<<(3 * Hv) / 8, 256, 0, stream>>>(W_ih, b_ih, emb_zt, ctx, gx);
  k_gemv_gh<<<(3 * Hv) / 8, 256, 0, stream>>>(W_hh, b_hh, last_h, gh);
  k_gru_cell<<<(Bv * Hv) / 256, 256, 0, stream>>>(gx, gh, last_h, emb_zt, out, st);

  // 4) scores (gen GEMV w/ async-LDS staging; copy WMMA 2560 waves + reduce)
  k_score_gen<<<Vv / 8, 256, 0, stream>>>(Wgen_w, Wgen_b, st, logits);
  k_copy_gemm<<<(NCH_CP * Bv * Tv / 16) / 4, 128, 0, stream>>>(encb, wcpub, wcpmb, wcpzb,
                                                               Wcpu_b, Wcpm_b, Wcppz_b, st, cppart);
  k_cp_reduce<<<(Bv * Tv) / 256, 256, 0, stream>>>(cppart, logits);

  // 5) softmax over 32320, mix, scatter
  k_softmax_big<<<Bv, 256, 0, stream>>>(logits);
  k_prob_out<<<dim3((Vv / 4 + 255) / 256, Bv), 256, 0, stream>>>(logits, pvz_prob, out);
  k_scatter<<<1, 64, 0, stream>>>(logits, u_input, m_input, out);
}
